// Game_14345190769036
// MI455X (gfx1250) — compile-verified, matrix operand order checked by reading, not run on hardware
//
#include <hip/hip_runtime.h>

#define MEMORY_LENGTH 32
#define N_TARGETS     3
#define T_STEPS       1000
#define DT            0.1f
#define HIDDEN        128
#define N_AGENTS      1024
#define IN_DIM        128          // 32*3 + 32
#define APB           16           // agents per block (one WMMA M-tile)
#define NBLK          (N_AGENTS / APB)   // 64 blocks

typedef _Float16 v8h  __attribute__((ext_vector_type(8)));
typedef _Float16 v16h __attribute__((ext_vector_type(16)));
typedef float    v8f  __attribute__((ext_vector_type(8)));

#if defined(__has_builtin) && __has_builtin(__builtin_amdgcn_sched_barrier)
#define SCHED_FENCE() __builtin_amdgcn_sched_barrier(0)
#else
#define SCHED_FENCE() __asm__ volatile("" ::: "memory")
#endif

__device__ __forceinline__ float fast_tanh(float x) {
    float cx = fminf(fmaxf(x, -15.0f), 15.0f);
    float e  = __expf(2.0f * cx);
    return __fdividef(e - 1.0f, e + 1.0f);
}

// ---------------------------------------------------------------------------
// Kernel 1: zero the global cost accumulators and convert W1 -> f16 in d_ws.
// ---------------------------------------------------------------------------
__global__ void prep_kernel(const float* __restrict__ W1,
                            _Float16* __restrict__ W1h,
                            float* __restrict__ acc) {
    int i = blockIdx.x * blockDim.x + threadIdx.x;
    if (i < 2) acc[i] = 0.0f;
    if (i < HIDDEN * IN_DIM) W1h[i] = (_Float16)W1[i];
}

// ---------------------------------------------------------------------------
// Kernel 2: persistent rollout. 64 blocks x 256 threads (8 wave32 per block),
// each block simulates 16 agents for all 1000 steps with LDS-resident state.
// Per step: wave w computes h-columns [16w,16w+16) with 4 x wmma_f32_16x16x32,
// then wave 0 does the W2 head (padded 16x16 WMMA) + unicycle update while
// waves 1-7 shift the feature history into the ping-pong buffer.
// ---------------------------------------------------------------------------
__global__ void __launch_bounds__(256, 1)
sim_kernel(const float* __restrict__ target_pos,   // [3][2]
           const float* __restrict__ logsigma,     // [3]
           const float* __restrict__ x_inits,      // [1024][3]
           const _Float16* __restrict__ W1h,       // [128][128] f16 (row: hidden)
           const float* __restrict__ b1,           // [128]
           const float* __restrict__ W2,           // [2][128]
           const float* __restrict__ b2,           // [2]
           float* __restrict__ acc) {              // [2] global accumulators
    __shared__ _Float16 feats[2][APB][IN_DIM];     // ping-pong feature ring  8KB
    __shared__ _Float16 hbuf[APB][HIDDEN];         // hidden activations      4KB
    __shared__ float    util[APB][16];             // W2 wmma output tile     1KB
    __shared__ float    posS[APB][3];              // init staging only
    __shared__ float    concS[APB];                // init staging only

    const int tid   = threadIdx.x;
    const int wave  = tid >> 5;                    // 0..7 -> N-tile id
    const int lane  = tid & 31;
    const int hl    = lane & 15;
    const int kbase = (lane < 16) ? 0 : 8;         // K striping of 16b A/B frags
    const int mbase = (lane < 16) ? 0 : 8;         // M striping of f32 C/D tile
    const int a0    = blockIdx.x * APB;

    // --- loop-invariant per-lane constants -------------------------------
    float tx[N_TARGETS], ty[N_TARGETS], invsig[N_TARGETS];
#pragma unroll
    for (int j = 0; j < N_TARGETS; ++j) {
        tx[j] = target_pos[2 * j];
        ty[j] = target_pos[2 * j + 1];
        invsig[j] = 1.0f / __expf(logsigma[j]);
    }
    const float b2v0 = b2[0], b2v1 = b2[1];
    const float b1v  = b1[wave * 16 + hl];

    // --- preload B fragments of W1 for this wave's 16-column N-tile ------
    // B[k][n] = W1[n][k]; lane n holds column (wave*16 + n), K striped as A.
    v16h bfrag[4];
    {
        const _Float16* wrow = W1h + (wave * 16 + hl) * IN_DIM;
#pragma unroll
        for (int kc = 0; kc < 4; ++kc) {
            const v8h* p0 = (const v8h*)(wrow + kc * 32 + kbase);
            const v8h* p1 = (const v8h*)(wrow + kc * 32 + kbase + 16);
            bfrag[kc] = __builtin_shufflevector(*p0, *p1,
                0, 1, 2, 3, 4, 5, 6, 7, 8, 9, 10, 11, 12, 13, 14, 15);
        }
    }
    // --- preload padded W2 B fragments (columns >= 2 are zero) -----------
    v16h b2frag[4];
#pragma unroll
    for (int kc = 0; kc < 4; ++kc) {
        v16h t;
#pragma unroll
        for (int e = 0; e < 16; ++e) {
            int k = kc * 32 + kbase + (e < 8 ? e : 8 + e);
            t[e] = (hl < 2) ? (_Float16)W2[hl * HIDDEN + k] : (_Float16)0.0f;
        }
        b2frag[kc] = t;
    }

    // --- initial state: wave0 lanes 0..15 own one agent each -------------
    float xr = 0.0f, yr = 0.0f, thr = 0.0f;        // agent state registers
    if (tid < APB) {
        xr  = x_inits[(a0 + tid) * 3 + 0];
        yr  = x_inits[(a0 + tid) * 3 + 1];
        thr = x_inits[(a0 + tid) * 3 + 2];
        posS[tid][0] = xr; posS[tid][1] = yr; posS[tid][2] = thr;
        float c = 0.0f;
#pragma unroll
        for (int j = 0; j < N_TARGETS; ++j) {
            float dx = xr - tx[j], dy = yr - ty[j];
            c += invsig[j] * __expf(-(dx * dx + dy * dy) * invsig[j]);
        }
        concS[tid] = c;
    }
    __syncthreads();
    for (int idx = tid; idx < APB * IN_DIM; idx += 256) {
        int a = idx >> 7, j = idx & 127;
        feats[0][a][j] = (j < 96) ? (_Float16)posS[a][j % 3] : (_Float16)concS[a];
    }
    __syncthreads();

    float costF = 0.0f;   // sum over t of -concentration(x_next)
    float costU = 0.0f;   // sum over t of v^2 + w^2
    int p = 0;

#pragma unroll 1
    for (int t = 0; t < T_STEPS; ++t) {
        // ---- h = relu(feats @ W1^T + b1): one 16x16 N-tile per wave -----
        // Issue all 8 ds_load_b128 back-to-back, then fence the scheduler so
        // the WMMA chain pays LDS latency once instead of four times.
        const _Float16* arow = &feats[p][hl][0];
        v16h afrag[4];
#pragma unroll
        for (int kc = 0; kc < 4; ++kc) {
            const v8h* q0 = (const v8h*)(arow + kc * 32 + kbase);
            const v8h* q1 = (const v8h*)(arow + kc * 32 + kbase + 16);
            afrag[kc] = __builtin_shufflevector(*q0, *q1,
                0, 1, 2, 3, 4, 5, 6, 7, 8, 9, 10, 11, 12, 13, 14, 15);
        }
        SCHED_FENCE();
        v8f cacc = {};
#pragma unroll
        for (int kc = 0; kc < 4; ++kc)
            cacc = __builtin_amdgcn_wmma_f32_16x16x32_f16(
                false, afrag[kc], false, bfrag[kc], (short)0, cacc, false, false);
        {
            const int col = wave * 16 + hl;
#pragma unroll
            for (int r = 0; r < 8; ++r) {
                float hv = cacc[r] + b1v;
                hbuf[mbase + r][col] = (_Float16)(hv > 0.0f ? hv : 0.0f);
            }
        }
        __syncthreads();   // hbuf ready; feats[p] reads for GEMM done

        if (wave == 0) {
            // ---- u = tanh(h @ W2^T + b2) via padded WMMA ----------------
            const _Float16* hrow = &hbuf[hl][0];
            v16h hfrag[4];
#pragma unroll
            for (int kc = 0; kc < 4; ++kc) {
                const v8h* q0 = (const v8h*)(hrow + kc * 32 + kbase);
                const v8h* q1 = (const v8h*)(hrow + kc * 32 + kbase + 16);
                hfrag[kc] = __builtin_shufflevector(*q0, *q1,
                    0, 1, 2, 3, 4, 5, 6, 7, 8, 9, 10, 11, 12, 13, 14, 15);
            }
            SCHED_FENCE();
            v8f c2 = {};
#pragma unroll
            for (int kc = 0; kc < 4; ++kc)
                c2 = __builtin_amdgcn_wmma_f32_16x16x32_f16(
                    false, hfrag[kc], false, b2frag[kc], (short)0, c2, false, false);
#pragma unroll
            for (int r = 0; r < 8; ++r) util[mbase + r][hl] = c2[r];
            __asm__ volatile("" ::: "memory");   // keep DS store->load order (HW in-order)

            if (lane < APB) {
                // ---- unicycle update + field eval (agent = lane) --------
                float v = fast_tanh(util[lane][0] + b2v0);
                float w = fast_tanh(util[lane][1] + b2v1);
                xr  += DT * v * __cosf(thr);
                yr  += DT * v * __sinf(thr);
                thr += DT * w;
                float c = 0.0f;
#pragma unroll
                for (int j = 0; j < N_TARGETS; ++j) {
                    float dx = xr - tx[j], dy = yr - ty[j];
                    c += invsig[j] * __expf(-(dx * dx + dy * dy) * invsig[j]);
                }
                // new head entries of next feature buffer
                feats[1 - p][lane][0]  = (_Float16)xr;
                feats[1 - p][lane][1]  = (_Float16)yr;
                feats[1 - p][lane][2]  = (_Float16)thr;
                feats[1 - p][lane][96] = (_Float16)c;
                costF -= c;               // sum_j -1/sigma exp(-d2/sigma)
                costU += v * v + w * w;
            }
        } else {
            // ---- waves 1..7: shift old history into next buffer ---------
            // (overlaps with wave 0's serial W2/update section)
            for (int idx = tid - 32; idx < APB * IN_DIM; idx += 224) {
                int a = idx >> 7, j = idx & 127;
                if (j >= 3 && j != 96)
                    feats[1 - p][a][j] = feats[p][a][j - ((j < 96) ? 3 : 1)];
            }
        }
        __syncthreads();   // next feature buffer complete
        p ^= 1;
    }

    if (wave == 0 && lane < APB) {
        atomicAdd(&acc[0], costF);
        atomicAdd(&acc[1], costU);
    }
}

// ---------------------------------------------------------------------------
// Kernel 3: combine accumulators into the scalar cost.
// ---------------------------------------------------------------------------
__global__ void fin_kernel(const float* __restrict__ acc, float* __restrict__ out) {
    const float nt = (float)N_AGENTS * (float)T_STEPS;
    out[0] = acc[0] / nt + acc[1] / (nt * 2.0f);
}

extern "C" void kernel_launch(void* const* d_in, const int* in_sizes, int n_in,
                              void* d_out, int out_size, void* d_ws, size_t ws_size,
                              hipStream_t stream) {
    const float* target_pos = (const float*)d_in[0];
    const float* logsigma   = (const float*)d_in[1];
    const float* x_inits    = (const float*)d_in[2];
    const float* W1         = (const float*)d_in[3];
    const float* b1         = (const float*)d_in[4];
    const float* W2         = (const float*)d_in[5];
    const float* b2         = (const float*)d_in[6];

    float*    acc = (float*)d_ws;                         // 2 accumulators
    _Float16* W1h = (_Float16*)((char*)d_ws + 256);       // 32 KB f16 weights

    prep_kernel<<<(HIDDEN * IN_DIM + 255) / 256, 256, 0, stream>>>(W1, W1h, acc);
    sim_kernel<<<NBLK, 256, 0, stream>>>(target_pos, logsigma, x_inits,
                                         W1h, b1, W2, b2, acc);
    fin_kernel<<<1, 1, 0, stream>>>(acc, (float*)d_out);
}
